// PointGroup_38534446580078
// MI455X (gfx1250) — compile-verified, matrix operand order checked by reading, not run on hardware
//
#include <hip/hip_runtime.h>
#include <hip/hip_bf16.h>
#include <stdint.h>

#define NCL 256
#define PTS 2048
#define FSV 14
#define NVOX (FSV * FSV * FSV) /* 2744 */
#define CF 32
#define SCALE_MAX 50.0f
#define THR 0.3f
#define GRID_ELEMS ((size_t)NCL * NVOX * CF)
#define FLTMAX 3.402823466e+38f

typedef __attribute__((ext_vector_type(4))) unsigned int u32x4;
typedef __attribute__((ext_vector_type(8))) unsigned int u32x8;
typedef __attribute__((ext_vector_type(8))) int v8i;

// ---------------------------------------------------------------------------
// Kernel 1: per-cluster stats (mean/min/max -> scale/offset) + per-point voxel
// id. clusters_idx slice staged into LDS via the Tensor Data Mover (async DMA).
// ---------------------------------------------------------------------------
__global__ __launch_bounds__(256) void pg_stats_voxid(
    const int* __restrict__ clusters_idx,
    const int* __restrict__ clusters_offset,
    const float* __restrict__ coords,
    const float* __restrict__ rand_a,
    const float* __restrict__ rand_b,
    int* __restrict__ voxid_out)
{
    __shared__ int   sidx[PTS * 2];   // 16 KB: (cid,pid) pairs for this cluster
    __shared__ float red[8][9];
    __shared__ float params[7];       // mean xyz, cscale, offset xyz

    const int c    = blockIdx.x;
    const int t    = threadIdx.x;
    const int off0 = clusters_offset[c];
    const int off1 = clusters_offset[c + 1];
    const int npts = off1 - off0;

    // ---- TDM async bulk copy: clusters_idx[2*off0 .. 2*off1) -> LDS --------
    if (t < 32) {
        unsigned n_dw = (unsigned)(npts * 2);                    // 4096 dwords
        unsigned long long ga =
            (unsigned long long)(const void*)(clusters_idx + (size_t)off0 * 2);
        unsigned lds_addr = (unsigned)(size_t)(void*)&sidx[0];
        u32x4 g0;
        g0.x = 1u;                                   // count=1 (valid D#)
        g0.y = lds_addr;                             // lds_addr
        g0.z = (unsigned)(ga & 0xFFFFFFFFu);         // global_addr[31:0]
        g0.w = (unsigned)((ga >> 32) & 0x1FFFFFFu)   // global_addr[56:32]
             | (2u << 30);                           // type=2 ("image")
        u32x8 g1;
        g1.s0 = (2u << 16);                          // data_size = 4 bytes
        g1.s1 = (n_dw & 0xFFFFu) << 16;              // tensor_dim0[15:0]
        g1.s2 = (n_dw >> 16) | (1u << 16);           // tensor_dim0[31:16] | tensor_dim1=1
        g1.s3 = (n_dw & 0xFFFFu) << 16;              // tile_dim0
        g1.s4 = 0u;                                  // tile_dim1=0, tile_dim2=0
        g1.s5 = n_dw;                                // tensor_dim0_stride[31:0]
        g1.s6 = 0u;
        g1.s7 = 0u;
        asm volatile("tensor_load_to_lds %0, %1" :: "s"(g0), "s"(g1) : "memory");
        __builtin_amdgcn_s_wait_tensorcnt(0);
    }
    __syncthreads();

    // ---- per-thread partials over 8 points ---------------------------------
    float sx = 0.f, sy = 0.f, sz = 0.f;
    float mnx = FLTMAX, mny = FLTMAX, mnz = FLTMAX;
    float mxx = -FLTMAX, mxy = -FLTMAX, mxz = -FLTMAX;
    for (int k = 0; k < 8; ++k) {
        int e = t + 256 * k;
        if (e < npts) {
            int pid = sidx[2 * e + 1];
            float x = coords[pid * 3 + 0];
            float y = coords[pid * 3 + 1];
            float z = coords[pid * 3 + 2];
            sx += x; sy += y; sz += z;
            mnx = fminf(mnx, x); mny = fminf(mny, y); mnz = fminf(mnz, z);
            mxx = fmaxf(mxx, x); mxy = fmaxf(mxy, y); mxz = fmaxf(mxz, z);
        }
    }
    // ---- wave32 reduction --------------------------------------------------
    for (int m = 16; m > 0; m >>= 1) {
        sx += __shfl_xor(sx, m); sy += __shfl_xor(sy, m); sz += __shfl_xor(sz, m);
        mnx = fminf(mnx, __shfl_xor(mnx, m));
        mny = fminf(mny, __shfl_xor(mny, m));
        mnz = fminf(mnz, __shfl_xor(mnz, m));
        mxx = fmaxf(mxx, __shfl_xor(mxx, m));
        mxy = fmaxf(mxy, __shfl_xor(mxy, m));
        mxz = fmaxf(mxz, __shfl_xor(mxz, m));
    }
    int wid = t >> 5, lid = t & 31;
    if (lid == 0) {
        red[wid][0] = sx;  red[wid][1] = sy;  red[wid][2] = sz;
        red[wid][3] = mnx; red[wid][4] = mny; red[wid][5] = mnz;
        red[wid][6] = mxx; red[wid][7] = mxy; red[wid][8] = mxz;
    }
    __syncthreads();
    if (t == 0) {
        float SX = 0.f, SY = 0.f, SZ = 0.f;
        float NX = FLTMAX, NY = FLTMAX, NZ = FLTMAX;
        float XX = -FLTMAX, XY = -FLTMAX, XZ = -FLTMAX;
        for (int w = 0; w < 8; ++w) {
            SX += red[w][0]; SY += red[w][1]; SZ += red[w][2];
            NX = fminf(NX, red[w][3]); NY = fminf(NY, red[w][4]); NZ = fminf(NZ, red[w][5]);
            XX = fmaxf(XX, red[w][6]); XY = fmaxf(XY, red[w][7]); XZ = fmaxf(XZ, red[w][8]);
        }
        float inv_n = 1.0f / (float)npts;
        float mx = SX * inv_n, my = SY * inv_n, mz = SZ * inv_n;
        float cminx = NX - mx, cminy = NY - my, cminz = NZ - mz;
        float cmaxx = XX - mx, cmaxy = XY - my, cmaxz = XZ - mz;
        float rmax = fmaxf(fmaxf((cmaxx - cminx), (cmaxy - cminy)), (cmaxz - cminz)) / (float)FSV;
        float cscale = fminf(1.0f / rmax - 0.01f, SCALE_MAX);
        float minx = cminx * cscale, miny = cminy * cscale, minz = cminz * cscale;
        float rngx = (cmaxx - cminx) * cscale;
        float rngy = (cmaxy - cminy) * cscale;
        float rngz = (cmaxz - cminz) * cscale;
        float ra0 = rand_a[0], ra1 = rand_a[1], ra2 = rand_a[2];
        float rb0 = rand_b[0], rb1 = rand_b[1], rb2 = rand_b[2];
        params[0] = mx; params[1] = my; params[2] = mz;
        params[3] = cscale;
        params[4] = -minx + fmaxf((float)FSV - rngx - 0.001f, 0.f) * ra0
                          + fminf((float)FSV - rngx + 0.001f, 0.f) * rb0;
        params[5] = -miny + fmaxf((float)FSV - rngy - 0.001f, 0.f) * ra1
                          + fminf((float)FSV - rngy + 0.001f, 0.f) * rb1;
        params[6] = -minz + fmaxf((float)FSV - rngz - 0.001f, 0.f) * ra2
                          + fminf((float)FSV - rngz + 0.001f, 0.f) * rb2;
    }
    __syncthreads();

    // ---- per-point voxel id, packed (vox<<18)|pid --------------------------
    float mx = params[0], my = params[1], mz = params[2];
    float cs = params[3];
    float ox = params[4], oy = params[5], oz = params[6];
    for (int k = 0; k < 8; ++k) {
        int e = t + 256 * k;
        if (e < npts) {
            int pid = sidx[2 * e + 1];
            float X = (coords[pid * 3 + 0] - mx) * cs + ox;
            float Y = (coords[pid * 3 + 1] - my) * cs + oy;
            float Z = (coords[pid * 3 + 2] - mz) * cs + oz;
            int vx = min(max((int)X, 0), FSV - 1);
            int vy = min(max((int)Y, 0), FSV - 1);
            int vz = min(max((int)Z, 0), FSV - 1);
            int v = (vx * FSV + vy) * FSV + vz;
            voxid_out[off0 + e] = (v << 18) | pid;
        }
    }
}

// ---------------------------------------------------------------------------
// Kernel 2: per-cluster voxel mean-pooling in LDS (ds_add_f32 atomics),
// 4 channels per pass x 8 passes; feats rows read as float4 (b128).
// ---------------------------------------------------------------------------
__global__ __launch_bounds__(256) void pg_pool(
    const int* __restrict__ voxid,
    const int* __restrict__ clusters_offset,
    const float* __restrict__ feats,
    float* __restrict__ out)
{
    __shared__ float acc[NVOX * 4];   // 43.9 KB
    __shared__ float cnt[NVOX];       // 11.0 KB
    const int c    = blockIdx.x;
    const int t    = threadIdx.x;
    const int off0 = clusters_offset[c];
    const int npts = clusters_offset[c + 1] - off0;
    const float4* f4 = (const float4*)feats;

    for (int p = 0; p < 8; ++p) {
        for (int i = t; i < NVOX * 4; i += 256) acc[i] = 0.f;
        if (p == 0)
            for (int i = t; i < NVOX; i += 256) cnt[i] = 0.f;
        __syncthreads();
        for (int k = 0; k < 8; ++k) {
            int e = t + 256 * k;
            if (e < npts) {
                int packed = voxid[off0 + e];
                int v   = packed >> 18;
                int pid = packed & 0x3FFFF;
                float4 f = f4[(size_t)pid * 8 + p];
                atomicAdd(&acc[v * 4 + 0], f.x);
                atomicAdd(&acc[v * 4 + 1], f.y);
                atomicAdd(&acc[v * 4 + 2], f.z);
                atomicAdd(&acc[v * 4 + 3], f.w);
                if (p == 0) atomicAdd(&cnt[v], 1.0f);
            }
        }
        __syncthreads();
        for (int v = t; v < NVOX; v += 256) {
            float inv = 1.0f / fmaxf(cnt[v], 1.0f);
            size_t base = ((size_t)(c * NVOX + v)) * CF + p * 4;
            out[base + 0] = acc[v * 4 + 0] * inv;
            out[base + 1] = acc[v * 4 + 1] * inv;
            out[base + 2] = acc[v * 4 + 2] * inv;
            out[base + 3] = acc[v * 4 + 3] * inv;
        }
        __syncthreads();
    }
}

// ---------------------------------------------------------------------------
// Kernel 3: NMS. Rank = row-sums of a 256x256 0/1 comparison matrix, computed
// as (cmp)x(ones) with v_wmma_i32_16x16x64_iu8; then sequential greedy sweep.
// ---------------------------------------------------------------------------
__global__ __launch_bounds__(256) void pg_nms(
    const float* __restrict__ ious,
    const float* __restrict__ scores,
    float* __restrict__ keep_out)
{
    __shared__ float sc[256];
    __shared__ int order[256];
    __shared__ int sup[256];
    const int t = threadIdx.x;
    sc[t]  = scores[t];
    sup[t] = 0;
    __syncthreads();

    const int lane = t & 31;
    const int wid  = t >> 5;
    const int h    = lane >> 4;   // which lane-half (K layout)

    v8i ones;
#pragma unroll
    for (int r = 0; r < 8; ++r) ones[r] = 0x01010101;

#pragma unroll
    for (int half = 0; half < 2; ++half) {
        const int rb  = half * 128 + wid * 16;     // 16-row tile base
        const int row = rb + (lane & 15);          // this lane's A-matrix row
        const float si = sc[row];
        v8i acc;
#pragma unroll
        for (int r = 0; r < 8; ++r) acc[r] = 0;
        for (int kt = 0; kt < 4; ++kt) {           // K = 4 x 64 = 256 cols
            v8i amat;
#pragma unroll
            for (int r = 0; r < 8; ++r) {
                // 8-bit A 16x64 layout: vgpr r covers K base below (4 bytes)
                int kb = ((r & 3) >> 1) * 16 + (r & 1) * 4 + h * 8 + (r >> 2) * 32
                       + kt * 64;
                unsigned d = 0;
#pragma unroll
                for (int b = 0; b < 4; ++b) {
                    int j = kb + b;
                    float sj = sc[j];
                    unsigned cm = ((sj > si) || (sj == si && j < row)) ? 1u : 0u;
                    d |= cm << (8 * b);
                }
                amat[r] = (int)d;
            }
            acc = __builtin_amdgcn_wmma_i32_16x16x64_iu8(
                false, amat, false, ones, acc, false, false);
        }
        // C/D layout: vgpr r, lanes 0-15 -> M=r ; lanes 16-31 -> M=r+8.
        // Column 0 lives in lanes 0 and 16; every column holds the row sum.
        if ((lane & 15) == 0) {
#pragma unroll
            for (int r = 0; r < 8; ++r) {
                int rk = acc[r];
                order[rk] = rb + h * 8 + r;        // order[rank] = index
            }
        }
    }
    __syncthreads();

    // Sequential greedy NMS over sorted order (matches reference fori_loop).
    for (int it = 0; it < 256; ++it) {
        int idx = order[it];
        int active = sup[idx] ? 0 : 1;
        __syncthreads();
        if (t == 0) keep_out[idx] = active ? 1.0f : 0.0f;
        if (active && ious[idx * 256 + t] > THR) sup[t] = 1;
        __syncthreads();
    }
}

// ---------------------------------------------------------------------------
extern "C" void kernel_launch(void* const* d_in, const int* in_sizes, int n_in,
                              void* d_out, int out_size, void* d_ws, size_t ws_size,
                              hipStream_t stream)
{
    const int*   clusters_idx    = (const int*)d_in[0];
    const int*   clusters_offset = (const int*)d_in[1];
    const float* feats           = (const float*)d_in[2];
    const float* coords          = (const float*)d_in[3];
    const float* ious            = (const float*)d_in[4];
    const float* scores          = (const float*)d_in[5];
    const float* rand_a          = (const float*)d_in[6];
    const float* rand_b          = (const float*)d_in[7];
    float* out   = (float*)d_out;
    int*   voxid = (int*)d_ws;   // NCL*PTS ints = 2 MB scratch

    pg_stats_voxid<<<NCL, 256, 0, stream>>>(clusters_idx, clusters_offset,
                                            coords, rand_a, rand_b, voxid);
    pg_pool<<<NCL, 256, 0, stream>>>(voxid, clusters_offset, feats, out);
    pg_nms<<<1, 256, 0, stream>>>(ious, scores, out + GRID_ELEMS);
}
